// MultiHeadAttentionWrapper_27608049778752
// MI455X (gfx1250) — matrix operand order repaired and checked by reading, not compiled, not run on hardware
//
#include <hip/hip_runtime.h>

// Problem constants (match reference)
#define B_   8
#define S_   2048
#define DIN  1024
#define H_   8
#define DH   128

typedef __attribute__((ext_vector_type(16))) __bf16 v16bf;
typedef __attribute__((ext_vector_type(8)))  float  v8f;

union ABu {
    v16bf v;
    uint4 u[2];
};

static __device__ __forceinline__ v8f wmma_bf16(const ABu& a, const ABu& b, v8f c) {
    return __builtin_amdgcn_wmma_f32_16x16x32_bf16(
        /*neg_a=*/false, a.v, /*neg_b=*/false, b.v,
        /*c_mod=*/(short)0, c, /*reuse_a=*/false, /*reuse_b=*/false);
}

// --- CDNA5 async global->LDS copy (ASYNCcnt path), with safe fallback -------
#if __has_builtin(__builtin_amdgcn_global_load_async_to_lds_b128) && \
    __has_builtin(__builtin_amdgcn_s_wait_asynccnt)
#define HAVE_ASYNC_LDS 1
#else
#define HAVE_ASYNC_LDS 0
#endif

#if HAVE_ASYNC_LDS
// Matches the builtin's parameter type reported by clang:
//   '__attribute__((__vector_size__(4 * sizeof(int)))) int __device__ *'
typedef int v4i_b128 __attribute__((vector_size(16)));
#endif

static __device__ __forceinline__ void async_copy_b128(const void* g, void* l) {
#if HAVE_ASYNC_LDS
    __builtin_amdgcn_global_load_async_to_lds_b128(
        (__attribute__((address_space(1))) v4i_b128*)g,
        (__attribute__((address_space(3))) v4i_b128*)l, 0, 0);
#else
    *(uint4*)l = *(const uint4*)g;
#endif
}

static __device__ __forceinline__ void async_wait_all() {
#if HAVE_ASYNC_LDS
    __builtin_amdgcn_s_wait_asynccnt(0);
#endif
}

// ---------------------------------------------------------------------------
// Kernel 1: fp32 -> bf16 conversion (grid-stride)
// ---------------------------------------------------------------------------
__global__ void f32_to_bf16(const float* __restrict__ in, __bf16* __restrict__ out, size_t n) {
    size_t i = (size_t)blockIdx.x * blockDim.x + threadIdx.x;
    size_t stride = (size_t)gridDim.x * blockDim.x;
    for (; i < n; i += stride) out[i] = (__bf16)in[i];
}

// ---------------------------------------------------------------------------
// Kernel 2: QKV projection GEMM.
//   For each (which in {q,k,v}, b, h): [S x DIN] @ [DIN x DH] -> [S x DH] (bf16 out)
//   Block: 256 threads (8 waves). Block tile: 128 rows x 128 cols.
//   Wave w computes rows [16w,16w+16) x all 128 cols = 8 WMMA tiles.
// ---------------------------------------------------------------------------
__global__ __launch_bounds__(256) void qkv_gemm(const __bf16* __restrict__ xb,
                                                const __bf16* __restrict__ wb,
                                                __bf16* __restrict__ qkv) {
    __shared__ __bf16 As[128 * 32];   // A tile, row-major [row][k], stride 32
    __shared__ __bf16 Bt[128 * 32];   // B tile, transposed [n][k], stride 32

    const int tid  = threadIdx.x;
    const int lane = tid & 31;
    const int wave = tid >> 5;
    const int row0 = blockIdx.x * 128;
    const int b    = blockIdx.y;
    const int z    = blockIdx.z;          // which*H + h
    const int which = z >> 3;
    const int h     = z & 7;

    const __bf16* X = xb + ((size_t)b * S_ + row0) * DIN;
    const __bf16* W = wb + ((size_t)which * H_ + h) * (size_t)DIN * DH;
    __bf16* out = qkv + (((size_t)which * B_ + b) * H_ + h) * (size_t)S_ * DH
                      + (size_t)row0 * DH;

    v8f acc[8];
#pragma unroll
    for (int i = 0; i < 8; i++) acc[i] = v8f{};

    const int kk8  = (lane >> 4) << 3;   // A-frag K base: 0 or 8
    const int kk16 = (lane >> 4) << 4;   // unused; B-frag also uses kk8

    for (int k0 = 0; k0 < DIN; k0 += 32) {
        // A tile 128x32 via async global->LDS: 512 b128 chunks, 2 per thread.
        {
#pragma unroll
            for (int i = 0; i < 2; i++) {
                int c = tid * 2 + i;
                int r = c >> 2, sub = (c & 3) << 3;
                async_copy_b128(X + (size_t)r * DIN + k0 + sub, As + r * 32 + sub);
            }
        }
        // B tile transposed (needs data transform -> manual): Bt[n][k].
        {
            int k = tid >> 3, n0 = (tid & 7) * 16;
            const __bf16* src = W + (size_t)(k0 + k) * DH + n0;
#pragma unroll
            for (int i = 0; i < 16; i++) Bt[(n0 + i) * 32 + k] = src[i];
        }
        async_wait_all();
        __syncthreads();

        // A fragment for this wave's 16 rows (16x32, bf16).
        ABu a;
        {
            int r = (wave << 4) + (lane & 15);
            a.u[0] = *(const uint4*)(As + r * 32 + kk8);        // K = kk8..kk8+7
            a.u[1] = *(const uint4*)(As + r * 32 + kk8 + 16);   // K = kk8+16..kk8+23
        }
        // Preload all 8 B fragments, then stream 8 back-to-back WMMAs.
        ABu bfr[8];
#pragma unroll
        for (int cb = 0; cb < 8; cb++) {
            int n = (cb << 4) + (lane & 15);
            bfr[cb].u[0] = *(const uint4*)(Bt + n * 32 + kk8);
            bfr[cb].u[1] = *(const uint4*)(Bt + n * 32 + kk8 + 16);
        }
#pragma unroll
        for (int cb = 0; cb < 8; cb++)
            acc[cb] = wmma_bf16(a, bfr[cb], acc[cb]);
        __syncthreads();
    }

    // Store bf16 result (C layout: lane l, vgpr r -> M = r + 8*(l/16), N = l%16).
#pragma unroll
    for (int cb = 0; cb < 8; cb++) {
#pragma unroll
        for (int r = 0; r < 8; r++) {
            int row = (wave << 4) + r + ((lane >> 4) << 3);
            int col = (cb << 4) + (lane & 15);
            out[(size_t)row * DH + col] = (__bf16)acc[cb][r];
        }
    }
}

// ---------------------------------------------------------------------------
// Kernel 3: causal flash attention.
//   Block = (b, h, 128 query rows); 8 waves, wave w owns rows [16w,16w+16).
//   Streams kv tiles of 128: K tile async-staged row-major in LDS, V tile
//   transposed in LDS, online softmax, O accumulated in fp32.
// ---------------------------------------------------------------------------
__global__ __launch_bounds__(256) void flash_attn(const __bf16* __restrict__ qkv,
                                                  float* __restrict__ out) {
    __shared__ __bf16 Ks[128 * 128];       // K tile, row-major [kv][e], 32 KB
    __shared__ __bf16 Vt[128 * 128];       // V tile transposed: [e][kv], 32 KB
    __shared__ __bf16 Pw[8][16 * 128];     // per-wave P scratch, 32 KB

    const int tid   = threadIdx.x;
    const int lane  = tid & 31;
    const int wave  = tid >> 5;
    const int qbase = blockIdx.x * 128;
    const int h     = blockIdx.y;
    const int b     = blockIdx.z;

    const size_t plane = ((size_t)b * H_ + h) * (size_t)S_ * DH;
    const size_t tsz   = (size_t)B_ * H_ * S_ * DH;
    const __bf16* Qp = qkv + plane;
    const __bf16* Kp = qkv + tsz + plane;
    const __bf16* Vp = qkv + 2 * tsz + plane;

    const int kk8  = (lane >> 4) << 3;
    const int kk16 = (lane >> 4) << 4;
    const float scale = 0.08838834764831845f;   // 1/sqrt(128)

    // Q fragments: 4x (16x32) covering E=128.
    ABu qf[4];
    {
        int qrow = qbase + (wave << 4) + (lane & 15);
#pragma unroll
        for (int f = 0; f < 4; f++) {
            qf[f].u[0] = *(const uint4*)(Qp + (size_t)qrow * DH + 32 * f + kk8);
            qf[f].u[1] = *(const uint4*)(Qp + (size_t)qrow * DH + 32 * f + kk8 + 16);
        }
    }

    float m[8], lsum[8];
    v8f o[8];
#pragma unroll
    for (int r = 0; r < 8; r++) { m[r] = -3.0e38f; lsum[r] = 0.0f; }
#pragma unroll
    for (int cb = 0; cb < 8; cb++) o[cb] = v8f{};

    const int ntiles = blockIdx.x + 1;   // causal: only tiles at/below diagonal
    for (int t = 0; t < ntiles; ++t) {
        const int kvbase = t * 128;

        // Prefetch next tile's K/V into cache hierarchy (global_prefetch_b8).
        if (t + 1 < ntiles) {
            __builtin_prefetch(Kp + (size_t)(kvbase + 128 + (tid >> 1)) * DH, 0, 1);
            __builtin_prefetch(Vp + (size_t)(kvbase + 128 + (tid >> 1)) * DH, 0, 1);
        }

        // K tile: flat 32 KB copy via async global->LDS (2048 b128 chunks).
        {
            const __bf16* src = Kp + (size_t)kvbase * DH;
#pragma unroll
            for (int i = 0; i < 8; i++) {
                int c = tid + (i << 8);
                async_copy_b128(src + (c << 3), Ks + (c << 3));
            }
        }
        // V tile transposed (data transform -> manual): Vt[e][kv].
        {
            int kv = tid >> 1, e0 = (tid & 1) * 64;
            const __bf16* src = Vp + (size_t)(kvbase + kv) * DH + e0;
#pragma unroll
            for (int i = 0; i < 64; i++) Vt[(e0 + i) * 128 + kv] = src[i];
        }
        async_wait_all();
        __syncthreads();

        // S = Q K^T for this wave's 16 rows x 128 kv cols (8 col-blocks).
        v8f sc[8];
#pragma unroll
        for (int cb = 0; cb < 8; cb++) {
            v8f s = v8f{};
            int krow = (cb << 4) + (lane & 15);
            ABu kf[4];
#pragma unroll
            for (int f = 0; f < 4; f++) {
                kf[f].u[0] = *(const uint4*)(Ks + (size_t)krow * DH + 32 * f + kk8);
                kf[f].u[1] = *(const uint4*)(Ks + (size_t)krow * DH + 32 * f + kk8 + 16);
            }
#pragma unroll
            for (int f = 0; f < 4; f++)
                s = wmma_bf16(qf[f], kf[f], s);
            sc[cb] = s;
        }

        // Causal mask on the diagonal tile.
        if (t == blockIdx.x) {
#pragma unroll
            for (int cb = 0; cb < 8; cb++) {
                int j = kvbase + (cb << 4) + (lane & 15);
#pragma unroll
                for (int r = 0; r < 8; r++) {
                    int i = qbase + (wave << 4) + r + ((lane >> 4) << 3);
                    if (j > i) sc[cb][r] = -3.0e38f;
                }
            }
        }

        // Online softmax update (per row; rows live in 16-lane halves).
#pragma unroll
        for (int r = 0; r < 8; r++) {
            float cand = sc[0][r];
#pragma unroll
            for (int cb = 1; cb < 8; cb++) cand = fmaxf(cand, sc[cb][r]);
            cand *= scale;
#pragma unroll
            for (int msk = 1; msk < 16; msk <<= 1)
                cand = fmaxf(cand, __shfl_xor(cand, msk, 32));
            float mnew = fmaxf(m[r], cand);
            float corr = __expf(m[r] - mnew);
            m[r] = mnew;

            float part = 0.0f;
#pragma unroll
            for (int cb = 0; cb < 8; cb++) {
                float p = __expf(sc[cb][r] * scale - mnew);
                sc[cb][r] = p;
                part += p;
            }
#pragma unroll
            for (int msk = 1; msk < 16; msk <<= 1)
                part += __shfl_xor(part, msk, 32);
            lsum[r] = lsum[r] * corr + part;
#pragma unroll
            for (int cb = 0; cb < 8; cb++) o[cb][r] *= corr;
        }

        // Repack P (C layout) -> bf16 A fragments via per-wave LDS scratch.
        __bf16* P = &Pw[wave][0];   // 16 x 128, row-major
#pragma unroll
        for (int cb = 0; cb < 8; cb++) {
#pragma unroll
            for (int r = 0; r < 8; r++) {
                P[(r + ((lane >> 4) << 3)) * 128 + (cb << 4) + (lane & 15)] =
                    (__bf16)sc[cb][r];
            }
        }
        ABu pf[4];
        {
            int pr = lane & 15;
#pragma unroll
            for (int f = 0; f < 4; f++) {
                pf[f].u[0] = *(const uint4*)(P + pr * 128 + 32 * f + kk8);
                pf[f].u[1] = *(const uint4*)(P + pr * 128 + 32 * f + kk8 + 16);
            }
        }

        // O += P @ V  (8 output e-blocks x 4 kv chunks).
#pragma unroll
        for (int cb = 0; cb < 8; cb++) {
            v8f oacc = o[cb];
            int e = (cb << 4) + (lane & 15);
            ABu vf[4];
#pragma unroll
            for (int f = 0; f < 4; f++) {
                vf[f].u[0] = *(const uint4*)(Vt + e * 128 + 32 * f + kk8);
                vf[f].u[1] = *(const uint4*)(Vt + e * 128 + 32 * f + kk8 + 16);
            }
#pragma unroll
            for (int f = 0; f < 4; f++)
                oacc = wmma_bf16(pf[f], vf[f], oacc);
            o[cb] = oacc;
        }
        __syncthreads();   // Ks/Vt reused next tile
    }

    // Final normalize + store fp32 to [B, S, H*DH].
#pragma unroll
    for (int cb = 0; cb < 8; cb++) {
#pragma unroll
        for (int r = 0; r < 8; r++) {
            int i = qbase + (wave << 4) + r + ((lane >> 4) << 3);
            int e = (cb << 4) + (lane & 15);
            out[((size_t)b * S_ + i) * (H_ * DH) + h * DH + e] = o[cb][r] / lsum[r];
        }
    }
}

// ---------------------------------------------------------------------------
// Launcher
// ---------------------------------------------------------------------------
extern "C" void kernel_launch(void* const* d_in, const int* in_sizes, int n_in,
                              void* d_out, int out_size, void* d_ws, size_t ws_size,
                              hipStream_t stream) {
    (void)in_sizes; (void)n_in; (void)out_size; (void)ws_size;
    const float* x  = (const float*)d_in[0];
    const float* Wq = (const float*)d_in[1];
    const float* Wk = (const float*)d_in[2];
    const float* Wv = (const float*)d_in[3];
    float* out = (float*)d_out;

    const size_t nX = (size_t)B_ * S_ * DIN;          // 16,777,216
    const size_t nW = (size_t)H_ * DIN * DH;          //  1,048,576 per tensor

    __bf16* xb  = reinterpret_cast<__bf16*>(d_ws);
    __bf16* wb  = xb + nX;                            // 3 weight tensors, contiguous
    __bf16* qkv = wb + 3 * nW;                        // [3][B][H][S][DH] bf16

    f32_to_bf16<<<2048, 256, 0, stream>>>(x,  xb,          nX);
    f32_to_bf16<<<512,  256, 0, stream>>>(Wq, wb,          nW);
    f32_to_bf16<<<512,  256, 0, stream>>>(Wk, wb + nW,     nW);
    f32_to_bf16<<<512,  256, 0, stream>>>(Wv, wb + 2 * nW, nW);

    qkv_gemm<<<dim3(S_ / 128, B_, 3 * H_), 256, 0, stream>>>(xb, wb, qkv);
    flash_attn<<<dim3(S_ / 128, H_, B_), 256, 0, stream>>>(qkv, out);
}
